// TopKExpertRouter_56160992362643
// MI455X (gfx1250) — matrix operand order repaired and checked by reading, not compiled, not run on hardware
//
#include <hip/hip_runtime.h>
#include <hip/hip_bf16.h>

typedef __attribute__((ext_vector_type(16))) _Float16 v16h;
typedef __attribute__((ext_vector_type(8)))  _Float16 v8h;
typedef __attribute__((ext_vector_type(8)))  float    v8f;

#define B_TOK 4096
#define DDIM  1024
#define HDIM  4096
#define ODIM  1024
#define NEXP  8
#define NSLOT (2 * B_TOK)        // total (token, expert) assignments
#define SLOT_PAD (NSLOT + 128)   // allow full-tile overrun reads
#define SA 40                    // padded LDS row stride in halfs (bank-conflict free)

// ---------------------------------------------------------------- utilities
__global__ void k_zero(float* __restrict__ p, int n) {
  int i = blockIdx.x * blockDim.x + threadIdx.x;
  int stride = gridDim.x * blockDim.x;
  for (; i < n; i += stride) p[i] = 0.f;
}

// ---------------------------------------------------------------- gating
// One wave32 per token: logits = x[b,:] @ Wg + bg, top-2, softmax over the 2.
__global__ void k_gate(const float* __restrict__ x, const float* __restrict__ Wg,
                       const float* __restrict__ bg, unsigned* __restrict__ counts,
                       int* __restrict__ tk_e, float* __restrict__ tk_w) {
  int wv = threadIdx.x >> 5;
  int l  = threadIdx.x & 31;
  int b  = blockIdx.x * 8 + wv;
  float acc[NEXP];
#pragma unroll
  for (int e = 0; e < NEXP; ++e) acc[e] = 0.f;
  const float* xr = x + (size_t)b * DDIM;
  for (int d = l; d < DDIM; d += 32) {
    float xv = xr[d];
    const float* wr = Wg + d * NEXP;
#pragma unroll
    for (int e = 0; e < NEXP; ++e) acc[e] += xv * wr[e];
  }
#pragma unroll
  for (int off = 16; off >= 1; off >>= 1) {
#pragma unroll
    for (int e = 0; e < NEXP; ++e) acc[e] += __shfl_xor(acc[e], off, 32);
  }
  if (l == 0) {
    float lg[NEXP];
#pragma unroll
    for (int e = 0; e < NEXP; ++e) lg[e] = acc[e] + bg[e];
    int e0 = 0; float l0 = lg[0];
#pragma unroll
    for (int e = 1; e < NEXP; ++e) if (lg[e] > l0) { l0 = lg[e]; e0 = e; }
    int e1 = (e0 == 0) ? 1 : 0; float l1 = -3.4e38f;
#pragma unroll
    for (int e = 0; e < NEXP; ++e) if (e != e0 && lg[e] > l1) { l1 = lg[e]; e1 = e; }
    float w0 = 1.f / (1.f + __expf(l1 - l0));  // softmax over {l0, l1}, l0 >= l1
    float w1 = 1.f - w0;
    tk_e[2 * b]     = e0;  tk_w[2 * b]     = w0;
    tk_e[2 * b + 1] = e1;  tk_w[2 * b + 1] = w1;
    atomicAdd(&counts[e0], 1u);
    atomicAdd(&counts[e1], 1u);
  }
}

// ---------------------------------------------------------------- scan
__global__ void k_scan(const unsigned* __restrict__ counts, unsigned* __restrict__ cursor,
                       unsigned* __restrict__ offs) {
  if (threadIdx.x == 0 && blockIdx.x == 0) {
    unsigned s = 0;
    for (int e = 0; e < NEXP; ++e) { offs[e] = s; cursor[e] = s; s += counts[e]; }
    offs[NEXP] = s;
  }
}

// ---------------------------------------------------------------- gather
__global__ void k_gather(const float* __restrict__ x, const int* __restrict__ tk_e,
                         const float* __restrict__ tk_w, unsigned* __restrict__ cursor,
                         int* __restrict__ slot_tok, float* __restrict__ slot_w,
                         _Float16* __restrict__ xg) {
  __shared__ int sslot;
  int i = blockIdx.x;                 // assignment index = token*2 + k
  if (threadIdx.x == 0) {
    int e = tk_e[i];
    int s = (int)atomicAdd(&cursor[e], 1u);
    sslot = s;
    slot_tok[s] = i >> 1;
    slot_w[s]   = tk_w[i];
  }
  __syncthreads();
  int s = sslot;
  const float* xr = x + (size_t)(i >> 1) * DDIM;
  _Float16* dst = xg + (size_t)s * DDIM;
  for (int j = threadIdx.x; j < DDIM; j += 128) dst[j] = (_Float16)xr[j];
}

// ---------------------------------------------------------------- weight cvt+transpose
// W[e][k][n] fp32  ->  Wt[e][n][k] f16  (k contiguous so GEMM B-tiles are 16B runs)
__global__ __launch_bounds__(256) void k_cvt_t(const float* __restrict__ W,
                                               _Float16* __restrict__ Wt,
                                               int K, int N) {
  __shared__ _Float16 t[32][36];
  int e  = blockIdx.z;
  int k0 = blockIdx.y * 32;
  int n0 = blockIdx.x * 32;
  const float* src = W + ((size_t)e * K + k0) * N + n0;
  int tn = threadIdx.x & 31;
  int tk = (threadIdx.x >> 5) * 4;
#pragma unroll
  for (int i = 0; i < 4; ++i) {
    int kk = tk + i;
    t[tn][kk] = (_Float16)src[(size_t)kk * N + tn];
  }
  __syncthreads();
  _Float16* dst = Wt + ((size_t)e * N + n0) * K + k0;
  int wn = threadIdx.x >> 3;
  int wk = (threadIdx.x & 7) * 4;
  *(uint2*)(dst + (size_t)wn * K + wk) = *(const uint2*)&t[wn][wk];
}

// ---------------------------------------------------------------- WMMA GEMM
// C[128,128] tile per block. A: f16 [rows,KTOT] compacted per expert.
// ASYNCP: B from pre-transposed f16 Wt[e][n][k]; A & B staged with
//         global_load_async_to_lds_b128, double-buffered LDS.
// else  : B from fp32 W[e][k][n], converted to f16 while staging (transposed).
// IS_G1 : epilogue relu(acc+b1) -> f16 H.  else (acc+b2)*slot_w atomicAdd out.
template <int KTOT, int NTOT, bool IS_G1, bool ASYNCP>
__global__ __launch_bounds__(256) void k_gemm(
    const _Float16* __restrict__ A, const float* __restrict__ W,
    const _Float16* __restrict__ Wt, const float* __restrict__ bias,
    const unsigned* __restrict__ offs, _Float16* __restrict__ Hout,
    const int* __restrict__ slot_tok, const float* __restrict__ slot_w,
    float* __restrict__ out) {
  __shared__ _Float16 As[2][128 * SA];
  __shared__ _Float16 Bt[2][128 * SA];

  int e  = blockIdx.y >> 5;
  int mt = blockIdx.y & 31;
  int rowbase = (int)offs[e] + mt * 128;
  int rowend  = (int)offs[e + 1];
  if (rowbase >= rowend) return;
  int n0  = blockIdx.x * 128;
  int tid = threadIdx.x;
  int w = tid >> 5, l = tid & 31;
  int mblk = w & 3;                   // rows [32*mblk, +32)
  int nblk = w >> 2;                  // cols [64*nblk, +64)
  int lm = l & 15, hsel = l >> 4;

  v8f acc[2][4];
#pragma unroll
  for (int mi = 0; mi < 2; ++mi)
#pragma unroll
    for (int ni = 0; ni < 4; ++ni) acc[mi][ni] = (v8f){0.f,0.f,0.f,0.f,0.f,0.f,0.f,0.f};

  const _Float16* Arow = A + (size_t)rowbase * KTOT;
  int ar = tid >> 1, ah = (tid & 1) * 16;   // A staging: 128 rows x 32 k, 32B/thread
  int bn = tid >> 1, bko = (tid & 1) * 16;  // B staging (async): 128 n x 32 k

#define STAGE_ASYNC(K0, BUF)                                                    \
  {                                                                             \
    const _Float16* _as = Arow + (size_t)ar * KTOT + (K0) + ah;                 \
    unsigned _ad = (unsigned)(uintptr_t)&As[(BUF)][ar * SA + ah];               \
    asm volatile("global_load_async_to_lds_b128 %0, %1, off"                    \
                 :: "v"(_ad), "v"(_as) : "memory");                             \
    asm volatile("global_load_async_to_lds_b128 %0, %1, off offset:16"         \
                 :: "v"(_ad), "v"(_as) : "memory");                             \
    const _Float16* _bs = Wtbase + (size_t)bn * KTOT + (K0) + bko;              \
    unsigned _bd = (unsigned)(uintptr_t)&Bt[(BUF)][bn * SA + bko];              \
    asm volatile("global_load_async_to_lds_b128 %0, %1, off"                    \
                 :: "v"(_bd), "v"(_bs) : "memory");                             \
    asm volatile("global_load_async_to_lds_b128 %0, %1, off offset:16"         \
                 :: "v"(_bd), "v"(_bs) : "memory");                             \
  }

  if constexpr (ASYNCP) {
    const _Float16* Wtbase = Wt + ((size_t)e * NTOT + n0) * KTOT;
    STAGE_ASYNC(0, 0);
    for (int k0 = 0; k0 < KTOT; k0 += 32) {
      int buf = (k0 >> 5) & 1;
      if (k0 + 32 < KTOT) {
        STAGE_ASYNC(k0 + 32, buf ^ 1);
        asm volatile("s_wait_asynccnt 0x4" ::: "memory");  // current buf done
      } else {
        asm volatile("s_wait_asynccnt 0x0" ::: "memory");
      }
      __syncthreads();
      const _Float16* Asb = As[buf];
      const _Float16* Btb = Bt[buf];
      v16h af[2];
#pragma unroll
      for (int mi = 0; mi < 2; ++mi) {
        int base = (mblk * 32 + mi * 16 + lm) * SA + hsel * 8;
        v8h lo = *(const v8h*)&Asb[base];
        v8h hi = *(const v8h*)&Asb[base + 16];
        af[mi] = __builtin_shufflevector(lo, hi, 0,1,2,3,4,5,6,7,8,9,10,11,12,13,14,15);
      }
#pragma unroll
      for (int ni = 0; ni < 4; ++ni) {
        int nb = (nblk * 64 + ni * 16 + lm) * SA + hsel * 8;
        v8h lo = *(const v8h*)&Btb[nb];
        v8h hi = *(const v8h*)&Btb[nb + 16];
        v16h bf = __builtin_shufflevector(lo, hi, 0,1,2,3,4,5,6,7,8,9,10,11,12,13,14,15);
#pragma unroll
        for (int mi = 0; mi < 2; ++mi)
          acc[mi][ni] = __builtin_amdgcn_wmma_f32_16x16x32_f16(
              false, af[mi], false, bf, (short)0, acc[mi][ni], false, false);
      }
      __syncthreads();
    }
  } else {
    const float* Wbase = W + (size_t)e * KTOT * NTOT + n0;
    int bk = tid >> 3, bq = (tid & 7) * 4;   // fallback B staging: fp32 -> f16 transpose
    for (int k0 = 0; k0 < KTOT; k0 += 32) {
      const _Float16* asrc = Arow + (size_t)ar * KTOT + k0 + ah;
      uint4 a0 = *(const uint4*)(asrc);
      uint4 a1 = *(const uint4*)(asrc + 8);
      *(uint4*)&As[0][ar * SA + ah]     = a0;
      *(uint4*)&As[0][ar * SA + ah + 8] = a1;
      const float* wsrc = Wbase + (size_t)(k0 + bk) * NTOT;
      __builtin_prefetch(wsrc + (size_t)32 * NTOT, 0, 1);
#pragma unroll
      for (int i = 0; i < 4; ++i) {
        int c = bq + i * 32;
        float4 f = *(const float4*)(wsrc + c);
        Bt[0][(c + 0) * SA + bk] = (_Float16)f.x;
        Bt[0][(c + 1) * SA + bk] = (_Float16)f.y;
        Bt[0][(c + 2) * SA + bk] = (_Float16)f.z;
        Bt[0][(c + 3) * SA + bk] = (_Float16)f.w;
      }
      __syncthreads();
      v16h af[2];
#pragma unroll
      for (int mi = 0; mi < 2; ++mi) {
        int base = (mblk * 32 + mi * 16 + lm) * SA + hsel * 8;
        v8h lo = *(const v8h*)&As[0][base];
        v8h hi = *(const v8h*)&As[0][base + 16];
        af[mi] = __builtin_shufflevector(lo, hi, 0,1,2,3,4,5,6,7,8,9,10,11,12,13,14,15);
      }
#pragma unroll
      for (int ni = 0; ni < 4; ++ni) {
        int nb = (nblk * 64 + ni * 16 + lm) * SA + hsel * 8;
        v8h lo = *(const v8h*)&Bt[0][nb];
        v8h hi = *(const v8h*)&Bt[0][nb + 16];
        v16h bf = __builtin_shufflevector(lo, hi, 0,1,2,3,4,5,6,7,8,9,10,11,12,13,14,15);
#pragma unroll
        for (int mi = 0; mi < 2; ++mi)
          acc[mi][ni] = __builtin_amdgcn_wmma_f32_16x16x32_f16(
              false, af[mi], false, bf, (short)0, acc[mi][ni], false, false);
      }
      __syncthreads();
    }
  }
#undef STAGE_ASYNC

  // ---- epilogue; D layout: m = r + 8*(lane/16), n = lane%16 ----
  const float* bptr = bias + (size_t)e * NTOT + n0;
#pragma unroll
  for (int mi = 0; mi < 2; ++mi) {
#pragma unroll
    for (int ni = 0; ni < 4; ++ni) {
      int col  = nblk * 64 + ni * 16 + lm;
      int gcol = n0 + col;
      float bv = bptr[col];
      int rb = rowbase + mblk * 32 + mi * 16 + hsel * 8;
      if (IS_G1) {
#pragma unroll
        for (int r = 0; r < 8; ++r) {
          int row = rb + r;
          if (row < rowend) {
            float v = acc[mi][ni][r] + bv;
            v = v > 0.f ? v : 0.f;
            Hout[(size_t)row * NTOT + gcol] = (_Float16)v;
          }
        }
      } else {
#pragma unroll
        for (int r = 0; r < 8; ++r) {
          int row = rb + r;
          if (row < rowend) {
            float v = acc[mi][ni][r] + bv;
            int tok  = slot_tok[row];
            float sw = slot_w[row];
            atomicAdd(out + (size_t)tok * ODIM + gcol, sw * v);
          }
        }
      }
    }
  }
}

// ---------------------------------------------------------------- launcher
extern "C" void kernel_launch(void* const* d_in, const int* in_sizes, int n_in,
                              void* d_out, int out_size, void* d_ws, size_t ws_size,
                              hipStream_t stream) {
  (void)in_sizes; (void)n_in; (void)out_size;
  const float* x  = (const float*)d_in[0];
  const float* Wg = (const float*)d_in[1];
  const float* bg = (const float*)d_in[2];
  const float* W1 = (const float*)d_in[3];
  const float* b1 = (const float*)d_in[4];
  const float* W2 = (const float*)d_in[5];
  const float* b2 = (const float*)d_in[6];
  float* out = (float*)d_out;

  char* ws = (char*)d_ws;
  unsigned* counts   = (unsigned*)(ws + 0);
  unsigned* cursor   = (unsigned*)(ws + 256);
  unsigned* offs     = (unsigned*)(ws + 512);
  int*      tk_e     = (int*)   (ws + 1024);
  float*    tk_w     = (float*) (ws + 1024 + (size_t)NSLOT * 4);
  int*      slot_tok = (int*)   (ws + 1024 + (size_t)NSLOT * 8);
  float*    slot_w   = (float*) (ws + 1024 + (size_t)NSLOT * 12);

  const size_t OFF_XG  = (size_t)1  << 20;             // xg : 8320*1024*2  ~ 16.3 MB
  const size_t OFF_HB  = (size_t)24 << 20;             // hb : 8320*4096*2  ~ 65 MB
  const size_t OFF_W1T = (size_t)96 << 20;             // W1t: 8*4096*1024*2 = 64 MB
  const size_t OFF_W2T = (size_t)164 << 20;            // W2t: 64 MB
  const size_t SZ_WT   = (size_t)NEXP * DDIM * HDIM * 2;
  _Float16* xg  = (_Float16*)(ws + OFF_XG);
  _Float16* hb  = (_Float16*)(ws + OFF_HB);
  _Float16* W1t = (_Float16*)(ws + OFF_W1T);
  _Float16* W2t = (_Float16*)(ws + OFF_W2T);
  bool use_async = ws_size >= OFF_W2T + SZ_WT;

  k_zero<<<4096, 256, 0, stream>>>(out, B_TOK * ODIM);
  k_zero<<<1, 32, 0, stream>>>((float*)counts, 8);
  if (use_async) {
    dim3 gc1(HDIM / 32, DDIM / 32, NEXP);   // W1: K=D, N=H
    k_cvt_t<<<gc1, 256, 0, stream>>>(W1, W1t, DDIM, HDIM);
    dim3 gc2(ODIM / 32, HDIM / 32, NEXP);   // W2: K=H, N=O
    k_cvt_t<<<gc2, 256, 0, stream>>>(W2, W2t, HDIM, ODIM);
  }
  k_gate<<<B_TOK / 8, 256, 0, stream>>>(x, Wg, bg, counts, tk_e, tk_w);
  k_scan<<<1, 32, 0, stream>>>(counts, cursor, offs);
  k_gather<<<NSLOT, 128, 0, stream>>>(x, tk_e, tk_w, cursor, slot_tok, slot_w, xg);

  dim3 g1(HDIM / 128, NEXP * 32);
  dim3 g2(ODIM / 128, NEXP * 32);
  if (use_async) {
    k_gemm<DDIM, HDIM, true,  true><<<g1, 256, 0, stream>>>(
        xg, nullptr, W1t, b1, offs, hb, nullptr, nullptr, nullptr);
    k_gemm<HDIM, ODIM, false, true><<<g2, 256, 0, stream>>>(
        hb, nullptr, W2t, b2, offs, nullptr, slot_tok, slot_w, out);
  } else {
    k_gemm<DDIM, HDIM, true,  false><<<g1, 256, 0, stream>>>(
        xg, W1, nullptr, b1, offs, hb, nullptr, nullptr, nullptr);
    k_gemm<HDIM, ODIM, false, false><<<g2, 256, 0, stream>>>(
        hb, W2, nullptr, b2, offs, nullptr, slot_tok, slot_w, out);
  }
}
